// GeometryLayer_18305150615723
// MI455X (gfx1250) — matrix operand order repaired and checked by reading, not compiled, not run on hardware
//
#include <hip/hip_runtime.h>
#include <hip/hip_bf16.h>
#include <stdint.h>
#include <stddef.h>

#ifndef __has_builtin
#define __has_builtin(x) 0
#endif

// ---------------------------------------------------------------------------
// Types
// ---------------------------------------------------------------------------
typedef _Float16 h16;
typedef h16          v16h  __attribute__((ext_vector_type(16)));
typedef float        v8f   __attribute__((ext_vector_type(8)));
typedef unsigned int u32x4 __attribute__((ext_vector_type(4)));
typedef int          i32x8 __attribute__((ext_vector_type(8)));
typedef int          i32x4 __attribute__((ext_vector_type(4)));

// Problem constants (match reference)
#define BB   4
#define CC   512
#define NN   1024
#define FF   2048
#define M1   (BB * NN)        // 4096 tokens  [B,N]
#define M2   (BB * 4 * NN)    // 16384 tokens [B,4,N] / [B,N,4]
#define EPS  1e-5f

#define LDT  40               // LDS row stride in f16 (64B data + 16B pad)

// ---------------------------------------------------------------------------
// WMMA fragment helpers (per CDNA5 ISA VGPR layouts, wave32)
// ---------------------------------------------------------------------------
union AFragU { v16h v; unsigned u[8]; };

__device__ __forceinline__ v8f wmma_f16(v16h a, v16h b, v8f c) {
  return __builtin_amdgcn_wmma_f32_16x16x32_f16(false, a, false, b, (short)0, c,
                                                false, false);
}

// A 16x32 f16: lanes 0-15 -> M=lane, lanes16-31 -> M=lane-16.
// VGPR v holds K pair: (v>=4?16:0) + (hiHalf?8:0) + (v&3)*2
__device__ __forceinline__ v16h load_frag_A(const h16* s, int row0) {
  const int lane = threadIdx.x & 31;
  const int m = lane & 15, hi = lane >> 4;
  const h16* base = s + (row0 + m) * LDT;
  AFragU f;
#pragma unroll
  for (int v = 0; v < 8; ++v) {
    const int kk = ((v & 4) << 2) + (hi << 3) + ((v & 3) << 1);
    f.u[v] = *(const unsigned*)(base + kk);
  }
  return f.v;
}

// B 32x16 f16: lanes hold column N=lane&15; VGPR v holds K pair (v>=4?16:0) + (hiHalf?8:0) + (v&3)*2
// (column j of B == row j of W[nOut,K], so we gather from W rows)
__device__ __forceinline__ v16h load_frag_B(const h16* s, int row0) {
  const int lane = threadIdx.x & 31;
  const int n = lane & 15, hi = lane >> 4;
  const h16* base = s + (row0 + n) * LDT;
  AFragU f;
#pragma unroll
  for (int v = 0; v < 8; ++v) {
    const int kk = ((v & 4) << 2) + (hi << 3) + ((v & 3) << 1);
    f.u[v] = *(const unsigned*)(base + kk);
  }
  return f.v;
}

// ---------------------------------------------------------------------------
// Tensor Data Mover: load one 64x32 f16 tile of W[nOut,K] into LDS with
// pad_enable so LDS row stride becomes 64B + 16B = 80B (= LDT f16).
// Descriptor layout per CDNA5 ISA.
// Toolchain probe: clang-23 lane => 6-arg builtin
//   (u32x4 g0, i32x8 g1, i32x4 g2, i32x4 g3, i32x8 g4, i32 cpol)
// ---------------------------------------------------------------------------
#if __has_builtin(__builtin_amdgcn_tensor_load_to_lds)
#define HAVE_TDM 1
__device__ __forceinline__ void tdm_load_tile_f16(const h16* gsrc,
                                                  unsigned lds_byte_addr,
                                                  int K, int rowsTotal) {
  const unsigned long long ga = (unsigned long long)(uintptr_t)gsrc;
  u32x4 g0;
  g0.x = 1u;                                      // count=1, user descriptor
  g0.y = lds_byte_addr;                           // lds_addr
  g0.z = (unsigned)ga;                            // global_addr[31:0]
  g0.w = (unsigned)((ga >> 32) & 0x1FFFFFFull)    // global_addr[56:32]
         | (2u << 30);                            // type = 2 ("image")
  i32x8 g1;
  // data_size=1 (2B), pad_enable, pad_interval=3 (16 dwords), pad_amount=3 (4 dwords)
  g1[0] = (int)((1u << 16) | (1u << 20) | (3u << 22) | (3u << 25));
  g1[1] = (int)(((unsigned)K & 0xFFFFu) << 16);                 // tensor_dim0 lo
  g1[2] = (int)((((unsigned)K >> 16) & 0xFFFFu)                 // tensor_dim0 hi
                | (((unsigned)rowsTotal & 0xFFFFu) << 16));     // tensor_dim1 lo
  g1[3] = (int)(((((unsigned)rowsTotal) >> 16) & 0xFFFFu)       // tensor_dim1 hi
                | (32u << 16));                                 // tile_dim0 = 32
  g1[4] = 64;                                                   // tile_dim1 = 64
  g1[5] = K;                                                    // dim0_stride lo
  g1[6] = 0;
  g1[7] = 0;
  i32x4 z4; z4[0] = 0; z4[1] = 0; z4[2] = 0; z4[3] = 0;
  i32x8 z8;
#pragma unroll
  for (int i = 0; i < 8; ++i) z8[i] = 0;
  __builtin_amdgcn_tensor_load_to_lds(g0, g1, z4, z4, z8, 0);
#if __has_builtin(__builtin_amdgcn_s_wait_tensorcnt)
  __builtin_amdgcn_s_wait_tensorcnt(0);
#endif
}
#else
#define HAVE_TDM 0
#endif

// ---------------------------------------------------------------------------
// Tiled WMMA GEMM:  out[M,Nout] = act( A[M,K] @ W[Nout,K]^T (+A2@W2^T) + bias )
// A, W are f16 row-major. M, Nout multiples of 64; K multiple of 32.
// Block 128 threads = 4 waves, tile 64x64, each wave 32x32 (2x2 fragments).
// ---------------------------------------------------------------------------
template <bool DUAL, bool RELU, bool SF32, bool SF16>
__global__ __launch_bounds__(128) void gemm_kernel(
    const h16* __restrict__ A, const h16* __restrict__ W,
    const float* __restrict__ bias, int K,
    const h16* __restrict__ A2, const h16* __restrict__ W2,
    const float* __restrict__ bias2, int K2,
    int M, int Nout, float* __restrict__ outF, h16* __restrict__ outH) {
  __shared__ __align__(16) h16 sA[64][LDT];
  __shared__ __align__(16) h16 sB[64][LDT];

  const int tid  = threadIdx.x;
  const int wave = tid >> 5;
  const int wm   = wave >> 1, wn = wave & 1;
  const int m0   = blockIdx.y << 6;
  const int n0   = blockIdx.x << 6;

  v8f zero = {0.f, 0.f, 0.f, 0.f, 0.f, 0.f, 0.f, 0.f};
  v8f acc[2][2];
  acc[0][0] = zero; acc[0][1] = zero; acc[1][0] = zero; acc[1][1] = zero;

  auto accumulate = [&](const h16* __restrict__ Ap, const h16* __restrict__ Wp,
                        int Kp) {
    for (int k0 = 0; k0 < Kp; k0 += 32) {
      // ---- stage A tile (64x32 f16) with 128-bit loads ----
#pragma unroll
      for (int p = 0; p < 2; ++p) {
        const int idx = (p << 7) + tid;
        const int row = idx >> 2;
        const int seg = (idx & 3) << 3;
        *(uint4*)&sA[row][seg] =
            *(const uint4*)(Ap + (size_t)(m0 + row) * Kp + k0 + seg);
      }
      if (k0 + 32 < Kp)   // hint next A tile toward the caches
        __builtin_prefetch(Ap + (size_t)(m0 + (tid >> 1)) * Kp + k0 + 32, 0, 1);

      // ---- stage B tile: W rows [n0, n0+64), K cols [k0, k0+32) ----
#if HAVE_TDM
      if (tid == 0) {
        tdm_load_tile_f16(Wp + (size_t)n0 * Kp + k0,
                          (unsigned)(uintptr_t)&sB[0][0], Kp, Nout);
      }
#else
#pragma unroll
      for (int p = 0; p < 2; ++p) {
        const int idx = (p << 7) + tid;
        const int row = idx >> 2;
        const int seg = (idx & 3) << 3;
        *(uint4*)&sB[row][seg] =
            *(const uint4*)(Wp + (size_t)(n0 + row) * Kp + k0 + seg);
      }
#endif
      __syncthreads();

      v16h af0 = load_frag_A(&sA[0][0], (wm << 5));
      v16h af1 = load_frag_A(&sA[0][0], (wm << 5) + 16);
      v16h bf0 = load_frag_B(&sB[0][0], (wn << 5));
      v16h bf1 = load_frag_B(&sB[0][0], (wn << 5) + 16);

      acc[0][0] = wmma_f16(af0, bf0, acc[0][0]);
      acc[0][1] = wmma_f16(af0, bf1, acc[0][1]);
      acc[1][0] = wmma_f16(af1, bf0, acc[1][0]);
      acc[1][1] = wmma_f16(af1, bf1, acc[1][1]);
      __syncthreads();
    }
  };

  accumulate(A, W, K);
  if (DUAL) accumulate(A2, W2, K2);

  // ---- epilogue: C/D layout is VGPR r, lane l -> (M=r+8*(l>=16), N=l&15) ----
  const int lane = tid & 31;
  const int ln = lane & 15, hi = lane >> 4;
#pragma unroll
  for (int fm = 0; fm < 2; ++fm) {
#pragma unroll
    for (int fn = 0; fn < 2; ++fn) {
      const int col = n0 + (wn << 5) + (fn << 4) + ln;
      float bv = bias[col];
      if (DUAL) bv += bias2[col];
#pragma unroll
      for (int r = 0; r < 8; ++r) {
        const int row = m0 + (wm << 5) + (fm << 4) + r + (hi << 3);
        float v = acc[fm][fn][r] + bv;
        if (RELU) v = fmaxf(v, 0.f);
        const size_t o = (size_t)row * Nout + col;
        if (SF32) outF[o] = v;
        if (SF16) outH[o] = (h16)v;
      }
    }
  }
}

// ---------------------------------------------------------------------------
// conv1x1(K<=16) + BN(running stats) + ReLU  -> f16 token-major [B*S, 512]
// in layout [B, K, S]
// ---------------------------------------------------------------------------
__global__ __launch_bounds__(256) void conv1_bn_relu_kernel(
    const float* __restrict__ in, const float* __restrict__ w1,
    const float* __restrict__ b1, const float* __restrict__ g,
    const float* __restrict__ beta, const float* __restrict__ mean,
    const float* __restrict__ var, h16* __restrict__ out, int K, int S) {
  const size_t e = (size_t)blockIdx.x * 256 + threadIdx.x;
  const int c = (int)(e & 511);
  const size_t t = e >> 9;
  const int b = (int)(t / S);
  const int s = (int)(t - (size_t)b * S);
  float acc = b1[c];
  for (int k = 0; k < K; ++k)
    acc += w1[c * K + k] * in[((size_t)b * K + k) * S + s];
  const float sc = g[c] * rsqrtf(var[c] + EPS);
  const float h = (acc - mean[c]) * sc + beta[c];
  out[e] = (h16)fmaxf(h, 0.f);
}

// f32 -> f16
__global__ __launch_bounds__(256) void cvt_f16_kernel(const float* __restrict__ in,
                                                      h16* __restrict__ out,
                                                      size_t n) {
  const size_t e = (size_t)blockIdx.x * 256 + threadIdx.x;
  if (e < n) out[e] = (h16)in[e];
}

// y[B,C,4,N] -> f16 tokens [B, 4N, C] with row = m*N + n
__global__ __launch_bounds__(256) void transpose_y_kernel(
    const float* __restrict__ y, h16* __restrict__ out) {
  const size_t e = (size_t)blockIdx.x * 256 + threadIdx.x;  // B*4N*C
  const int c = (int)(e & 511);
  const size_t tr = e >> 9;
  const int n = (int)(tr & 1023);
  const int mn = (int)(tr >> 10);
  const int m = mn & 3, b = mn >> 2;
  out[e] = (h16)y[(((size_t)b * CC + c) * 4 + m) * NN + n];
}

// x1 = gemm(pos)+gemm(cls) (+biases already) + x^T  -> f32 + f16 [B*N, C]
__global__ __launch_bounds__(256) void embed_combine_kernel(
    const float* __restrict__ gsum, const float* __restrict__ x,
    float* __restrict__ outF, h16* __restrict__ outH) {
  const size_t e = (size_t)blockIdx.x * 256 + threadIdx.x;  // M1*C
  const int c = (int)(e & 511);
  const size_t t = e >> 9;
  const int n = (int)(t & 1023);
  const int b = (int)(t >> 10);
  const float v = gsum[e] + x[((size_t)b * CC + c) * NN + n];
  outF[e] = v;
  outH[e] = (h16)v;
}

// gated cross attention core: one wave per token. v*g == k*k (faithful).
__global__ __launch_bounds__(128) void attn_kernel(const h16* __restrict__ q,
                                                   const h16* __restrict__ k,
                                                   h16* __restrict__ ctx) {
  const int wave = threadIdx.x >> 5, lane = threadIdx.x & 31;
  const int t = blockIdx.x * 4 + wave;   // < M1
  const int b = t >> 10, n = t & 1023;
  const h16* qrow = q + (size_t)t * CC;
  const h16* kbase = k + (size_t)b * (4 * NN) * CC;

  // phase 1: lane = h*4+m computes logits over HD=64
  const int h = lane >> 2, m = lane & 3;
  const h16* krow = kbase + (size_t)(4 * n + m) * CC + h * 64;
  const h16* qh = qrow + h * 64;
  float s = 0.f;
#pragma unroll 8
  for (int d = 0; d < 64; ++d) s += (float)qh[d] * (float)krow[d];
  s *= 0.125f;  // HD^-0.5
  float mx = fmaxf(s, __shfl_xor(s, 1));
  mx = fmaxf(mx, __shfl_xor(mx, 2));
  const float ex = __expf(s - mx);
  float sum = ex + __shfl_xor(ex, 1);
  sum += __shfl_xor(sum, 2);
  const float a = ex / sum;

  // phase 2: lane covers 16 output channels; out = sum_m attn * k^2
  const int c0 = lane << 4;
  const int h2 = c0 >> 6;
  float o[16];
#pragma unroll
  for (int j = 0; j < 16; ++j) o[j] = 0.f;
#pragma unroll
  for (int mm = 0; mm < 4; ++mm) {
    const float am = __shfl(a, h2 * 4 + mm);
    const h16* kr = kbase + (size_t)(4 * n + mm) * CC + c0;
#pragma unroll
    for (int j = 0; j < 16; ++j) {
      const float kv = (float)kr[j];
      o[j] += am * kv * kv;
    }
  }
  h16* dst = ctx + (size_t)t * CC + c0;
#pragma unroll
  for (int j = 0; j < 16; ++j) dst[j] = (h16)o[j];
}

// LayerNorm over C=512, one wave per row. Optional +res, ReLU-after-LN,
// f32/f16 stores, and transposed store to [B,C,N,4] (row = (b*N+n)*4+m).
template <bool ADDRES, bool RELU, bool SF32, bool SF16, bool TRANSOUT>
__global__ __launch_bounds__(128) void ln_kernel(
    const float* __restrict__ in, const float* __restrict__ res,
    const float* __restrict__ g, const float* __restrict__ beta,
    float* __restrict__ outF, h16* __restrict__ outH) {
  const int wave = threadIdx.x >> 5, lane = threadIdx.x & 31;
  const int r = blockIdx.x * 4 + wave;
  const float* row = in + (size_t)r * CC;
  float v[16];
  float sum = 0.f;
#pragma unroll
  for (int i = 0; i < 16; ++i) {
    float x = row[i * 32 + lane];
    if (ADDRES) x += res[(size_t)r * CC + i * 32 + lane];
    v[i] = x;
    sum += x;
  }
#pragma unroll
  for (int off = 16; off > 0; off >>= 1) sum += __shfl_xor(sum, off);
  const float mu = sum * (1.f / CC);
  float vs = 0.f;
#pragma unroll
  for (int i = 0; i < 16; ++i) {
    const float d = v[i] - mu;
    vs += d * d;
  }
#pragma unroll
  for (int off = 16; off > 0; off >>= 1) vs += __shfl_xor(vs, off);
  const float rs = rsqrtf(vs * (1.f / CC) + EPS);
#pragma unroll
  for (int i = 0; i < 16; ++i) {
    const int c = i * 32 + lane;
    float y = (v[i] - mu) * rs * g[c] + beta[c];
    if (RELU) y = fmaxf(y, 0.f);
    if (SF32) {
      if (TRANSOUT) {
        const int m = r & 3, bn = r >> 2;
        const int n = bn & 1023, b = bn >> 10;
        outF[((((size_t)b * CC + c) * NN + n) << 2) + m] = y;
      } else {
        outF[(size_t)r * CC + c] = y;
      }
    }
    if (SF16) outH[(size_t)r * CC + c] = (h16)y;
  }
}

// sigmoid(mgate[b,n,:8] @ gw[2048,8]^T + gb)  -> f32 [B*N, 2048]
__global__ __launch_bounds__(256) void gate_kernel(
    const float* __restrict__ mg, const float* __restrict__ gw,
    const float* __restrict__ gb, float* __restrict__ out) {
  const size_t e = (size_t)blockIdx.x * 256 + threadIdx.x;  // M1*2048
  const int j = (int)(e & 2047);
  const size_t t = e >> 11;
  float v = gb[j];
#pragma unroll
  for (int k = 0; k < 8; ++k) v += gw[j * 8 + k] * mg[t * 8 + k];
  out[e] = 1.f / (1.f + __expf(-v));
}

// yb f16 tokens [ (b*N+n)*4+m , c ] = feat_c[b,n,c] + rel[(b*4+m)*N+n, c]
__global__ __launch_bounds__(256) void yb_combine_kernel(
    const float* __restrict__ featc, const float* __restrict__ rel,
    h16* __restrict__ out) {
  const size_t e = (size_t)blockIdx.x * 256 + threadIdx.x;  // M2*C
  const int c = (int)(e & 511);
  const size_t t2 = e >> 9;
  const int m = (int)(t2 & 3);
  const size_t bn = t2 >> 2;
  const int n = (int)(bn & 1023);
  const int b = (int)(bn >> 10);
  const size_t tr = (((size_t)b * 4 + m) * NN + n);
  out[e] = (h16)(featc[bn * CC + c] + rel[tr * CC + c]);
}

// pre = y^T + feat_m * gate  (token order (b,n,m))
__global__ __launch_bounds__(256) void merge_combine_kernel(
    const float* __restrict__ y, const float* __restrict__ featm,
    const float* __restrict__ gate, float* __restrict__ out) {
  const size_t e = (size_t)blockIdx.x * 256 + threadIdx.x;  // M2*C
  const int c = (int)(e & 511);
  const size_t t2 = e >> 9;
  const int m = (int)(t2 & 3);
  const size_t bn = t2 >> 2;
  const int n = (int)(bn & 1023);
  const int b = (int)(bn >> 10);
  const float yv = y[(((size_t)b * CC + c) * 4 + m) * NN + n];
  out[e] = yv + featm[e] * gate[bn * 2048 + (size_t)m * CC + c];
}

// ---------------------------------------------------------------------------
// Host-side orchestration
// ---------------------------------------------------------------------------
static inline dim3 grid1d(size_t n) { return dim3((unsigned)((n + 255) / 256)); }

extern "C" void kernel_launch(void* const* d_in, const int* in_sizes, int n_in,
                              void* d_out, int out_size, void* d_ws,
                              size_t ws_size, hipStream_t stream) {
  // inputs (setup_inputs insertion order, params flattened recursively)
  const float* x     = (const float*)d_in[0];   // [B,C,N]
  const float* pos   = (const float*)d_in[1];   // [B,2,1,N]
  const float* cls   = (const float*)d_in[2];   // [B,10,1,N]
  const float* y     = (const float*)d_in[3];   // [B,C,4,N]
  // d_in[4] = gate (dead in reference: overwritten by k)
  const float* mgate = (const float*)d_in[5];   // [B,N,8]
  const float* relin = (const float*)d_in[6];   // [B,2,4,N]
  // params.pos (7..14), .cls (15..22), .rel (23..30)
  const float* pw1 = (const float*)d_in[7],  *pb1 = (const float*)d_in[8];
  const float* pbg = (const float*)d_in[9],  *pbb = (const float*)d_in[10];
  const float* pbm = (const float*)d_in[11], *pbv = (const float*)d_in[12];
  const float* pw2 = (const float*)d_in[13], *pb2 = (const float*)d_in[14];
  const float* cw1 = (const float*)d_in[15], *cb1 = (const float*)d_in[16];
  const float* cbg = (const float*)d_in[17], *cbb = (const float*)d_in[18];
  const float* cbm = (const float*)d_in[19], *cbv = (const float*)d_in[20];
  const float* cw2 = (const float*)d_in[21], *cb2 = (const float*)d_in[22];
  const float* rw1 = (const float*)d_in[23], *rb1 = (const float*)d_in[24];
  const float* rbg = (const float*)d_in[25], *rbb = (const float*)d_in[26];
  const float* rbm = (const float*)d_in[27], *rbv = (const float*)d_in[28];
  const float* rw2 = (const float*)d_in[29], *rb2 = (const float*)d_in[30];
  // params.attn (31..48)
  const float* wq = (const float*)d_in[31], *bq = (const float*)d_in[32];
  const float* wk = (const float*)d_in[33], *bk = (const float*)d_in[34];
  const float* wo = (const float*)d_in[39], *bo = (const float*)d_in[40];
  const float* a_l1w = (const float*)d_in[41], *a_l1b = (const float*)d_in[42];
  const float* a_l2w = (const float*)d_in[43], *a_l2b = (const float*)d_in[44];
  const float* a_ln2g = (const float*)d_in[45], *a_ln2b = (const float*)d_in[46];
  const float* a_ln3g = (const float*)d_in[47], *a_ln3b = (const float*)d_in[48];
  // params.merge (49..64)
  const float* p1w = (const float*)d_in[49], *p1b = (const float*)d_in[50];
  const float* plng = (const float*)d_in[51], *plnb = (const float*)d_in[52];
  const float* p2w = (const float*)d_in[53], *p2b = (const float*)d_in[54];
  const float* gw  = (const float*)d_in[55], *gb  = (const float*)d_in[56];
  const float* m_l1w = (const float*)d_in[57], *m_l1b = (const float*)d_in[58];
  const float* m_l2w = (const float*)d_in[59], *m_l2b = (const float*)d_in[60];
  const float* m_ln2g = (const float*)d_in[61], *m_ln2b = (const float*)d_in[62];
  const float* m_ln3g = (const float*)d_in[63], *m_ln3b = (const float*)d_in[64];

  float* out = (float*)d_out;  // [B,C,N,4]

  // ---- workspace bump allocator ----
  char* base = (char*)d_ws;
  size_t off = 0;
  auto alloc = [&](size_t bytes) {
    void* p = base + off;
    off = (off + bytes + 255) & ~((size_t)255);
    return p;
  };
  const size_t CW = (size_t)CC * CC;   // 512*512
  const size_t FW = (size_t)FF * CC;   // 2048*512

  h16* hposH = (h16*)alloc((size_t)M1 * CC * 2);
  h16* hclsH = (h16*)alloc((size_t)M1 * CC * 2);
  h16* hrelH = (h16*)alloc((size_t)M2 * CC * 2);
  // f16 weights
  h16* pw2H = (h16*)alloc(CW * 2);
  h16* cw2H = (h16*)alloc(CW * 2);
  h16* rw2H = (h16*)alloc(CW * 2);
  h16* wqH  = (h16*)alloc(CW * 2);
  h16* wkH  = (h16*)alloc(CW * 2);
  h16* woH  = (h16*)alloc(CW * 2);
  h16* al1H = (h16*)alloc(FW * 2);
  h16* al2H = (h16*)alloc(FW * 2);
  h16* p1wH = (h16*)alloc(CW * 2);
  h16* p2wH = (h16*)alloc(CW * 2);
  h16* ml1H = (h16*)alloc(FW * 2);
  h16* ml2H = (h16*)alloc(FW * 2);
  // activations
  float* tmpAF = (float*)alloc((size_t)M1 * CC * 4);  // embed gemm / feat / ff2
  float* x1F   = (float*)alloc((size_t)M1 * CC * 4);
  h16*   x1H   = (h16*)alloc((size_t)M1 * CC * 2);
  h16*   ykH   = (h16*)alloc((size_t)M2 * CC * 2);
  h16*   kH    = (h16*)alloc((size_t)M2 * CC * 2);
  h16*   qH    = (h16*)alloc((size_t)M1 * CC * 2);
  h16*   ctxH  = (h16*)alloc((size_t)M1 * CC * 2);
  float* hF    = (float*)alloc((size_t)M1 * CC * 4);
  h16*   hH    = (h16*)alloc((size_t)M1 * CC * 2);
  h16*   ff1H  = (h16*)alloc((size_t)M1 * FF * 2);
  float* featcF = (float*)alloc((size_t)M1 * CC * 4);
  float* big1F = (float*)alloc((size_t)M2 * CC * 4);  // rel out, then pre
  h16*   ybH   = (h16*)alloc((size_t)M2 * CC * 2);
  float* big2F = (float*)alloc((size_t)M2 * CC * 4);  // p1 out, then gate
  h16*   hmH   = (h16*)alloc((size_t)M2 * CC * 2);
  float* big3F = (float*)alloc((size_t)M2 * CC * 4);  // feat_m, then mff2
  float* h2F   = (float*)alloc((size_t)M2 * CC * 4);
  h16*   h2H   = (h16*)alloc((size_t)M2 * CC * 2);
  h16*   mff1H = (h16*)alloc((size_t)M2 * FF * 2);
  (void)ws_size; (void)n_in; (void)in_sizes; (void)out_size;

  // ---- 0) f16 weight copies ----
  cvt_f16_kernel<<<grid1d(CW), 256, 0, stream>>>(pw2, pw2H, CW);
  cvt_f16_kernel<<<grid1d(CW), 256, 0, stream>>>(cw2, cw2H, CW);
  cvt_f16_kernel<<<grid1d(CW), 256, 0, stream>>>(rw2, rw2H, CW);
  cvt_f16_kernel<<<grid1d(CW), 256, 0, stream>>>(wq, wqH, CW);
  cvt_f16_kernel<<<grid1d(CW), 256, 0, stream>>>(wk, wkH, CW);
  cvt_f16_kernel<<<grid1d(CW), 256, 0, stream>>>(wo, woH, CW);
  cvt_f16_kernel<<<grid1d(FW), 256, 0, stream>>>(a_l1w, al1H, FW);
  cvt_f16_kernel<<<grid1d(FW), 256, 0, stream>>>(a_l2w, al2H, FW);
  cvt_f16_kernel<<<grid1d(CW), 256, 0, stream>>>(p1w, p1wH, CW);
  cvt_f16_kernel<<<grid1d(CW), 256, 0, stream>>>(p2w, p2wH, CW);
  cvt_f16_kernel<<<grid1d(FW), 256, 0, stream>>>(m_l1w, ml1H, FW);
  cvt_f16_kernel<<<grid1d(FW), 256, 0, stream>>>(m_l2w, ml2H, FW);

  // ---- 1) small convs + BN + ReLU ----
  conv1_bn_relu_kernel<<<grid1d((size_t)M1 * CC), 256, 0, stream>>>(
      pos, pw1, pb1, pbg, pbb, pbm, pbv, hposH, 2, NN);
  conv1_bn_relu_kernel<<<grid1d((size_t)M1 * CC), 256, 0, stream>>>(
      cls, cw1, cb1, cbg, cbb, cbm, cbv, hclsH, 10, NN);
  conv1_bn_relu_kernel<<<grid1d((size_t)M2 * CC), 256, 0, stream>>>(
      relin, rw1, rb1, rbg, rbb, rbm, rbv, hrelH, 2, 4 * NN);

  // ---- 2) embed: dual GEMM (pos conv2 + cls conv2), then + x^T ----
  gemm_kernel<true, false, true, false>
      <<<dim3(CC / 64, M1 / 64), 128, 0, stream>>>(
          hposH, pw2H, pb2, CC, hclsH, cw2H, cb2, CC, M1, CC, tmpAF, nullptr);
  embed_combine_kernel<<<grid1d((size_t)M1 * CC), 256, 0, stream>>>(tmpAF, x,
                                                                    x1F, x1H);

  // ---- 3) attention projections ----
  transpose_y_kernel<<<grid1d((size_t)M2 * CC), 256, 0, stream>>>(y, ykH);
  gemm_kernel<false, false, false, true>
      <<<dim3(CC / 64, M2 / 64), 128, 0, stream>>>(
          ykH, wkH, bk, CC, nullptr, nullptr, nullptr, 0, M2, CC, nullptr, kH);
  gemm_kernel<false, false, false, true>
      <<<dim3(CC / 64, M1 / 64), 128, 0, stream>>>(
          x1H, wqH, bq, CC, nullptr, nullptr, nullptr, 0, M1, CC, nullptr, qH);

  // ---- 4) attention core + out proj + LN ----
  attn_kernel<<<M1 / 4, 128, 0, stream>>>(qH, kH, ctxH);
  gemm_kernel<false, false, true, false>
      <<<dim3(CC / 64, M1 / 64), 128, 0, stream>>>(
          ctxH, woH, bo, CC, nullptr, nullptr, nullptr, 0, M1, CC, tmpAF,
          nullptr);
  ln_kernel<true, false, true, true, false><<<M1 / 4, 128, 0, stream>>>(
      tmpAF, x1F, a_ln2g, a_ln2b, hF, hH);

  // ---- 5) attn FFN + LN -> feat_c ----
  gemm_kernel<false, true, false, true>
      <<<dim3(FF / 64, M1 / 64), 128, 0, stream>>>(
          hH, al1H, a_l1b, CC, nullptr, nullptr, nullptr, 0, M1, FF, nullptr,
          ff1H);
  gemm_kernel<false, false, true, false>
      <<<dim3(CC / 64, M1 / 64), 128, 0, stream>>>(
          ff1H, al2H, a_l2b, FF, nullptr, nullptr, nullptr, 0, M1, CC, tmpAF,
          nullptr);
  ln_kernel<true, false, true, false, false><<<M1 / 4, 128, 0, stream>>>(
      tmpAF, hF, a_ln3g, a_ln3b, featcF, nullptr);

  // ---- 6) rel conv2 + yb = feat_c + rel ----
  gemm_kernel<false, false, true, false>
      <<<dim3(CC / 64, M2 / 64), 128, 0, stream>>>(
          hrelH, rw2H, rb2, CC, nullptr, nullptr, nullptr, 0, M2, CC, big1F,
          nullptr);
  yb_combine_kernel<<<grid1d((size_t)M2 * CC), 256, 0, stream>>>(featcF, big1F,
                                                                 ybH);

  // ---- 7) merge: p1 -> relu(LN) -> p2, gate, combine, LN ----
  gemm_kernel<false, false, true, false>
      <<<dim3(CC / 64, M2 / 64), 128, 0, stream>>>(
          ybH, p1wH, p1b, CC, nullptr, nullptr, nullptr, 0, M2, CC, big2F,
          nullptr);
  ln_kernel<false, true, false, true, false><<<M2 / 4, 128, 0, stream>>>(
      big2F, nullptr, plng, plnb, nullptr, hmH);
  gemm_kernel<false, false, true, false>
      <<<dim3(CC / 64, M2 / 64), 128, 0, stream>>>(
          hmH, p2wH, p2b, CC, nullptr, nullptr, nullptr, 0, M2, CC, big3F,
          nullptr);
  gate_kernel<<<grid1d((size_t)M1 * 2048), 256, 0, stream>>>(mgate, gw, gb,
                                                             big2F);
  merge_combine_kernel<<<grid1d((size_t)M2 * CC), 256, 0, stream>>>(
      y, big3F, big2F, big1F);
  ln_kernel<false, false, true, true, false><<<M2 / 4, 128, 0, stream>>>(
      big1F, nullptr, m_ln2g, m_ln2b, h2F, h2H);

  // ---- 8) merge FFN + LN(transposed store) -> d_out [B,C,N,4] ----
  gemm_kernel<false, true, false, true>
      <<<dim3(FF / 64, M2 / 64), 128, 0, stream>>>(
          h2H, ml1H, m_l1b, CC, nullptr, nullptr, nullptr, 0, M2, FF, nullptr,
          mff1H);
  gemm_kernel<false, false, true, false>
      <<<dim3(CC / 64, M2 / 64), 128, 0, stream>>>(
          mff1H, ml2H, m_l2b, FF, nullptr, nullptr, nullptr, 0, M2, CC, big3F,
          nullptr);
  ln_kernel<true, false, true, false, true><<<M2 / 4, 128, 0, stream>>>(
      big3F, h2F, m_ln3g, m_ln3b, out, nullptr);
}
